// Tree_Net_29643864277518
// MI455X (gfx1250) — matrix operand order repaired and checked by reading, not compiled, not run on hardware
//
#include <hip/hip_runtime.h>
#include <hip/hip_bf16.h>
#include <math.h>

typedef __attribute__((ext_vector_type(16))) _Float16 v16h;
typedef __attribute__((ext_vector_type(8)))  _Float16 v8h;
typedef __attribute__((ext_vector_type(8)))  float    v8f;

// ---------------------------------------------------------------------------
// f16 x f16 -> f32 GEMM using CDNA5 WMMA (v_wmma_f32_16x16x32_f16).
// Cout[M,N] = alpha * A[M,K] @ B[K,N] + beta * Cout + bias[N]
// Block tile 64x128, K-step 32, 256 threads = 8 waves (2x4).
// Each wave computes a 32x32 patch: 2 A-frags x 2 B-frags -> 4 WMMAs/K-step.
// A tile is staged with GLOBAL_LOAD_ASYNC_TO_LDS_B128 (ASYNCcnt-tracked DMA,
// no VGPR round trip); B tile is staged sync because it is transposed in LDS.
// ---------------------------------------------------------------------------
#define GEMM_BM 64
#define GEMM_BN 128
#define GEMM_BK 32
#define LDS_PAD 8   // halves; row stride 40 halves = 80 B (16B aligned)

__global__ __launch_bounds__(256) void gemm_wmma_f16(
    const _Float16* __restrict__ A, int lda,
    const _Float16* __restrict__ B, int ldb,
    float* __restrict__ Cout, int ldc,
    int M, int N, int K,
    float alpha, float beta,
    const float* __restrict__ bias)
{
    __shared__ _Float16 As [GEMM_BM][GEMM_BK + LDS_PAD];  // row-major A tile
    __shared__ _Float16 BsT[GEMM_BN][GEMM_BK + LDS_PAD];  // transposed B tile

    const int tid   = threadIdx.x;
    const int lane  = tid & 31;
    const int wave  = tid >> 5;       // 0..7
    const int waveM = wave >> 2;      // 0..1  (32-row slabs)
    const int waveN = wave & 3;       // 0..3  (32-col slabs)

    const int bM = blockIdx.x * GEMM_BM;
    const int bN = blockIdx.y * GEMM_BN;

    v8f acc00 = {}, acc01 = {}, acc10 = {}, acc11 = {};

    // A staging coords: 64 rows x 32 halves, one 16B chunk per thread
    const int arow = tid >> 2;            // 0..63
    const int acol = (tid & 3) * 8;       // 0,8,16,24
    // On gfx1250 the low 32 bits of a generic LDS address are the LDS offset.
    const unsigned a_ldsoff = (unsigned)(size_t)(&As[arow][acol]);
    const bool a_in_range   = (bM + arow) < M;
    const unsigned long long a_gbase =
        (unsigned long long)(size_t)(A + (size_t)(bM + arow) * lda + acol);

    // B staging coords: 32 k-rows x 128 n-cols, two v8h per thread
    const int bkrow0 = tid >> 4;                 // chunk 0: 0..15
    const int bncol0 = (tid & 15) * 8;
    const int bkrow1 = 16 + (tid >> 4);          // chunk 1: 16..31
    const int bncol1 = bncol0;

    for (int k0 = 0; k0 < K; k0 += GEMM_BK) {
        // ---- stage A tile: async DMA global->LDS (16B per lane) ----
        if (a_in_range) {
            const unsigned long long ga = a_gbase + (unsigned long long)k0 * 2u;
            asm volatile("global_load_async_to_lds_b128 %0, %1, off"
                         :: "v"(a_ldsoff), "v"(ga) : "memory");
        } else {
            *(v8h*)&As[arow][acol] = (v8h){};
        }
        // ---- stage B tile transposed (2 vector loads, scatter to LDS) ----
        {
            v8h bv0 = {}, bv1 = {};
            if (k0 + bkrow0 < K)
                bv0 = *(const v8h*)(B + (size_t)(k0 + bkrow0) * ldb + (bN + bncol0));
            if (k0 + bkrow1 < K)
                bv1 = *(const v8h*)(B + (size_t)(k0 + bkrow1) * ldb + (bN + bncol1));
            #pragma unroll
            for (int i = 0; i < 8; ++i) BsT[bncol0 + i][bkrow0] = bv0[i];
            #pragma unroll
            for (int i = 0; i < 8; ++i) BsT[bncol1 + i][bkrow1] = bv1[i];
        }
        // prefetch next K tile into L2 while we compute
        if (k0 + GEMM_BK < K) {
            if (a_in_range)
                __builtin_prefetch(A + (size_t)(bM + arow) * lda + (k0 + GEMM_BK + acol), 0, 1);
            __builtin_prefetch(B + (size_t)(k0 + GEMM_BK + bkrow0) * ldb + (bN + bncol0), 0, 1);
        }
        // drain the async LDS DMA before the tile barrier
        asm volatile("s_wait_asynccnt 0x0" ::: "memory");
        __syncthreads();

        // ---- A fragments: 16x32 f16; lane holds row lane%16,
        //      halves 0..7 = K (lane<16?0:8)+i ; halves 8..15 = +16
        const int rA  = lane & 15;
        const int kb0 = (lane < 16) ? 0 : 8;
        v16h afrag0, afrag1;
        {
            const _Float16* a0 = &As[waveM * 32 + rA][0];
            const _Float16* a1 = &As[waveM * 32 + 16 + rA][0];
            v8h lo0 = *(const v8h*)(a0 + kb0), hi0 = *(const v8h*)(a0 + kb0 + 16);
            v8h lo1 = *(const v8h*)(a1 + kb0), hi1 = *(const v8h*)(a1 + kb0 + 16);
            afrag0 = __builtin_shufflevector(lo0, hi0, 0,1,2,3,4,5,6,7,8,9,10,11,12,13,14,15);
            afrag1 = __builtin_shufflevector(lo1, hi1, 0,1,2,3,4,5,6,7,8,9,10,11,12,13,14,15);
        }
        // ---- B fragments: 32x16 f16; lane holds col lane%16,
        //      halves 0..7 = K (lane<16?0:8)+i ; halves 8..15 = +16
        const int cB  = lane & 15;
        const int kbB = (lane < 16) ? 0 : 8;
        v16h bfrag0, bfrag1;
        {
            const _Float16* b0 = &BsT[waveN * 32 + cB][kbB];
            const _Float16* b1 = &BsT[waveN * 32 + 16 + cB][kbB];
            v8h lo0 = *(const v8h*)(b0), hi0 = *(const v8h*)(b0 + 16);
            v8h lo1 = *(const v8h*)(b1), hi1 = *(const v8h*)(b1 + 16);
            bfrag0 = __builtin_shufflevector(lo0, hi0, 0,1,2,3,4,5,6,7,8,9,10,11,12,13,14,15);
            bfrag1 = __builtin_shufflevector(lo1, hi1, 0,1,2,3,4,5,6,7,8,9,10,11,12,13,14,15);
        }

        acc00 = __builtin_amdgcn_wmma_f32_16x16x32_f16(false, afrag0, false, bfrag0, (short)0, acc00, false, false);
        acc01 = __builtin_amdgcn_wmma_f32_16x16x32_f16(false, afrag0, false, bfrag1, (short)0, acc01, false, false);
        acc10 = __builtin_amdgcn_wmma_f32_16x16x32_f16(false, afrag1, false, bfrag0, (short)0, acc10, false, false);
        acc11 = __builtin_amdgcn_wmma_f32_16x16x32_f16(false, afrag1, false, bfrag1, (short)0, acc11, false, false);
        __syncthreads();
    }

    // ---- epilogue: f32 C/D layout: VGPR r -> row r + (lane<16?0:8), col lane%16
    const int col   = lane & 15;
    const int rbase = (lane < 16) ? 0 : 8;
    const int gm0   = bM + waveM * 32;
    const int gn0   = bN + waveN * 32;

    #pragma unroll
    for (int r = 0; r < 8; ++r) {
        const int r0 = gm0 + rbase + r;        // afrag0 rows
        const int r1 = r0 + 16;                // afrag1 rows
        const int c0 = gn0 + col;              // bfrag0 cols
        const int c1 = c0 + 16;                // bfrag1 cols
        if (r0 < M) {
            if (c0 < N) {
                float v = alpha * acc00[r];
                if (beta != 0.f) v += beta * Cout[(size_t)r0 * ldc + c0];
                if (bias) v += bias[c0];
                Cout[(size_t)r0 * ldc + c0] = v;
            }
            if (c1 < N) {
                float v = alpha * acc01[r];
                if (beta != 0.f) v += beta * Cout[(size_t)r0 * ldc + c1];
                if (bias) v += bias[c1];
                Cout[(size_t)r0 * ldc + c1] = v;
            }
        }
        if (r1 < M) {
            if (c0 < N) {
                float v = alpha * acc10[r];
                if (beta != 0.f) v += beta * Cout[(size_t)r1 * ldc + c0];
                if (bias) v += bias[c0];
                Cout[(size_t)r1 * ldc + c0] = v;
            }
            if (c1 < N) {
                float v = alpha * acc11[r];
                if (beta != 0.f) v += beta * Cout[(size_t)r1 * ldc + c1];
                if (bias) v += bias[c1];
                Cout[(size_t)r1 * ldc + c1] = v;
            }
        }
    }
}

// ---------------------------------------------------------------------------
// DFT cos/sin tables (f16):  cosM[j*D+k] = cos(2*pi*j*k/D), sinM = sin(...)
// ---------------------------------------------------------------------------
__global__ void init_dft_tables(_Float16* __restrict__ cosM,
                                _Float16* __restrict__ sinM, int D)
{
    size_t idx = (size_t)blockIdx.x * blockDim.x + threadIdx.x;
    size_t total = (size_t)D * D;
    if (idx >= total) return;
    int j = (int)(idx / D);
    int k = (int)(idx % D);
    int m = (int)(((long long)j * (long long)k) % D);   // exact phase mod D
    float ang = 6.28318530717958647692f * (float)m / (float)D;
    cosM[idx] = (_Float16)cosf(ang);
    sinM[idx] = (_Float16)sinf(ang);
}

// WT[k*C + n] = W[n*D + k]  (f32 [C,D] -> f16 [D,C])
__global__ void init_wt(const float* __restrict__ W, _Float16* __restrict__ WT,
                        int C, int D)
{
    size_t idx = (size_t)blockIdx.x * blockDim.x + threadIdx.x;
    size_t total = (size_t)D * C;
    if (idx >= total) return;
    int k = (int)(idx / C);
    int n = (int)(idx % C);
    WT[idx] = (_Float16)W[(size_t)n * D + k];
}

// leaves: vecs[node] = emb[word] ; also f16 copy
__global__ void gather_leaves(const int* __restrict__ info,
                              const float* __restrict__ emb,
                              float* __restrict__ vecs,
                              _Float16* __restrict__ vecsH,
                              int D)
{
    const int i    = blockIdx.x;
    const int node = info[2 * i + 0];
    const int word = info[2 * i + 1];
    for (int d = threadIdx.x; d < D; d += blockDim.x) {
        float v = emb[(size_t)word * D + d];
        vecs [(size_t)node * D + d] = v;
        vecsH[(size_t)node * D + d] = (_Float16)v;
    }
}

// ---------------------------------------------------------------------------
// Sequential chain in frequency domain. One block of D threads.
// C_p = conj(C_l) * C_r ; scale by rsqrt(sum|C|^2 / D)  (Parseval L2 norm)
// ---------------------------------------------------------------------------
__global__ __launch_bounds__(1024) void chain_freq(
    const int* __restrict__ comp,   // [steps,3] = l,r,p
    float* __restrict__ Fre, float* __restrict__ Fim,
    _Float16* __restrict__ FreH, _Float16* __restrict__ FimH,
    int steps, int D, int L)
{
    __shared__ float red[32];
    __shared__ float scale_sh;
    const int t    = threadIdx.x;      // 0..D-1
    const int lane = t & 31;
    const int wv   = t >> 5;

    for (int s = 0; s < steps; ++s) {
        const int l = comp[3 * s + 0];
        const int r = comp[3 * s + 1];
        const int p = comp[3 * s + 2];

        const float are = Fre[(size_t)l * D + t];
        const float aim = Fim[(size_t)l * D + t];
        const float bre = Fre[(size_t)r * D + t];
        const float bim = Fim[(size_t)r * D + t];
        // conj(A)*B
        const float cre = are * bre + aim * bim;
        const float cim = are * bim - aim * bre;

        float sq = cre * cre + cim * cim;
        #pragma unroll
        for (int off = 16; off > 0; off >>= 1)
            sq += __shfl_down(sq, off, 32);
        if (lane == 0) red[wv] = sq;
        __syncthreads();
        if (wv == 0) {
            float v = red[lane];
            #pragma unroll
            for (int off = 16; off > 0; off >>= 1)
                v += __shfl_down(v, off, 32);
            if (lane == 0) scale_sh = rsqrtf(v / (float)D + 1e-30f);
        }
        __syncthreads();
        const float sc  = scale_sh;
        const float ore = cre * sc;
        const float oim = cim * sc;
        Fre[(size_t)p * D + t] = ore;
        Fim[(size_t)p * D + t] = oim;
        const size_t hq = (size_t)(p - L) * D + t;
        FreH[hq] = (_Float16)ore;
        FimH[hq] = (_Float16)oim;
        __threadfence();
        __syncthreads();
    }
}

__global__ void f32_to_f16(const float* __restrict__ src,
                           _Float16* __restrict__ dst, size_t n)
{
    size_t i = (size_t)blockIdx.x * blockDim.x + threadIdx.x;
    if (i < n) dst[i] = (_Float16)src[i];
}

// ---------------------------------------------------------------------------
// Row softmax in place, one block per row, C <= 1024
// ---------------------------------------------------------------------------
__global__ __launch_bounds__(256) void softmax_rows(float* __restrict__ logits, int C)
{
    __shared__ float buf[1024];
    __shared__ float red[8];
    __shared__ float sval;
    const int row  = blockIdx.x;
    float* p = logits + (size_t)row * C;
    const int t = threadIdx.x, lane = t & 31, wv = t >> 5;

    float m = -3.4e38f;
    for (int i = t; i < C; i += 256) { float v = p[i]; buf[i] = v; m = fmaxf(m, v); }
    #pragma unroll
    for (int off = 16; off > 0; off >>= 1) m = fmaxf(m, __shfl_down(m, off, 32));
    if (lane == 0) red[wv] = m;
    __syncthreads();
    if (t == 0) {
        float mm = red[0];
        for (int i = 1; i < 8; ++i) mm = fmaxf(mm, red[i]);
        sval = mm;
    }
    __syncthreads();
    const float mx = sval;

    float s = 0.f;
    for (int i = t; i < C; i += 256) { float e = expf(buf[i] - mx); buf[i] = e; s += e; }
    #pragma unroll
    for (int off = 16; off > 0; off >>= 1) s += __shfl_down(s, off, 32);
    __syncthreads();                 // everyone done reading mx/red
    if (lane == 0) red[wv] = s;
    __syncthreads();
    if (t == 0) {
        float ss = 0.f;
        for (int i = 0; i < 8; ++i) ss += red[i];
        sval = ss;
    }
    __syncthreads();
    const float inv = 1.0f / sval;
    for (int i = t; i < C; i += 256) p[i] = buf[i] * inv;
}

// ---------------------------------------------------------------------------
static inline size_t align_up256(size_t x) { return (x + 255) & ~(size_t)255; }

extern "C" void kernel_launch(void* const* d_in, const int* in_sizes, int n_in,
                              void* d_out, int out_size, void* d_ws, size_t ws_size,
                              hipStream_t stream)
{
    (void)n_in; (void)out_size; (void)ws_size;

    const int*   leaf_info = (const int*)  d_in[0];   // [L,2]
    const int*   comp      = (const int*)  d_in[1];   // [L-1,3]
    const float* emb       = (const float*)d_in[2];   // [V,D]
    const float* lin_w     = (const float*)d_in[3];   // [C,D]
    const float* lin_b     = (const float*)d_in[4];   // [C]

    const int L = in_sizes[0] / 2;
    const int C = in_sizes[4];
    const int D = in_sizes[3] / C;
    const int N = 2 * L - 1;
    const int steps = L - 1;

    // workspace carve-out
    char* ws = (char*)d_ws;
    size_t off = 0;
    auto take = [&](size_t bytes) { char* p = ws + off; off = align_up256(off + bytes); return p; };

    _Float16* cosM  = (_Float16*)take((size_t)D * D * 2);
    _Float16* sinM  = (_Float16*)take((size_t)D * D * 2);
    _Float16* WT    = (_Float16*)take((size_t)D * C * 2);
    float*    vecsF = (float*)   take((size_t)N * D * 4);
    _Float16* vecsH = (_Float16*)take((size_t)N * D * 2);
    float*    Fre   = (float*)   take((size_t)N * D * 4);
    float*    Fim   = (float*)   take((size_t)N * D * 4);
    _Float16* FreH  = (_Float16*)take((size_t)steps * D * 2);
    _Float16* FimH  = (_Float16*)take((size_t)steps * D * 2);

    float* logits = (float*)d_out;   // [N, C]

    // 1) tables
    {
        size_t total = (size_t)D * D;
        init_dft_tables<<<dim3((unsigned)((total + 255) / 256)), dim3(256), 0, stream>>>(cosM, sinM, D);
        size_t totw = (size_t)D * C;
        init_wt<<<dim3((unsigned)((totw + 255) / 256)), dim3(256), 0, stream>>>(lin_w, WT, C, D);
    }
    // 2) leaves
    gather_leaves<<<dim3(L), dim3(256), 0, stream>>>(leaf_info, emb, vecsF, vecsH, D);

    // 3) forward DFT of leaves (WMMA GEMMs):  Fre = X@cos ; Fim = -(X@sin)
    {
        dim3 grid((L + GEMM_BM - 1) / GEMM_BM, (D + GEMM_BN - 1) / GEMM_BN);
        gemm_wmma_f16<<<grid, dim3(256), 0, stream>>>(vecsH, D, cosM, D, Fre, D,
                                                      L, D, D, 1.0f, 0.0f, nullptr);
        gemm_wmma_f16<<<grid, dim3(256), 0, stream>>>(vecsH, D, sinM, D, Fim, D,
                                                      L, D, D, -1.0f, 0.0f, nullptr);
    }
    // 4) sequential composition chain in frequency domain
    chain_freq<<<dim3(1), dim3(D), 0, stream>>>(comp, Fre, Fim, FreH, FimH, steps, D, L);

    // 5) inverse DFT (real part) for internal nodes:
    //    v = (Cre@cos)/D - (Cim@sin)/D   -> rows L..N-1 of vecsF
    {
        dim3 grid((steps + GEMM_BM - 1) / GEMM_BM, (D + GEMM_BN - 1) / GEMM_BN);
        float invD = 1.0f / (float)D;
        gemm_wmma_f16<<<grid, dim3(256), 0, stream>>>(FreH, D, cosM, D,
                                                      vecsF + (size_t)L * D, D,
                                                      steps, D, D, invD, 0.0f, nullptr);
        gemm_wmma_f16<<<grid, dim3(256), 0, stream>>>(FimH, D, sinM, D,
                                                      vecsF + (size_t)L * D, D,
                                                      steps, D, D, -invD, 1.0f, nullptr);
    }
    // 6) convert internal node vectors to f16
    {
        size_t n = (size_t)steps * D;
        f32_to_f16<<<dim3((unsigned)((n + 255) / 256)), dim3(256), 0, stream>>>(
            vecsF + (size_t)L * D, vecsH + (size_t)L * D, n);
    }
    // 7) logits GEMM with bias:  logits = vecs @ W.T + b
    {
        dim3 grid((N + GEMM_BM - 1) / GEMM_BM, (C + GEMM_BN - 1) / GEMM_BN);
        gemm_wmma_f16<<<grid, dim3(256), 0, stream>>>(vecsH, D, WT, C, logits, C,
                                                      N, C, D, 1.0f, 0.0f, lin_b);
    }
    // 8) softmax rows
    softmax_rows<<<dim3(N), dim3(256), 0, stream>>>(logits, C);
}
